// decoder_74972949119043
// MI455X (gfx1250) — compile-verified, matrix operand order checked by reading, not run on hardware
//
#include <hip/hip_runtime.h>
#include <hip/hip_bf16.h>

// ---------------------------------------------------------------------------
// 3-layer LSTM (T=32768, 128 -> 256 -> 512 -> 1) for MI455X (gfx1250).
//
// Strategy: the recurrence is latency/bandwidth bound, not FLOP bound.
// 16 persistent workgroups each keep a 1/16 shard of ALL weights (bf16,
// pre-swizzled into WMMA A-fragment layout) resident in LDS (256 KB of the
// 320 KB/WGP). Sharding is by h-index so each block owns rows of all four
// gates for its h slice and updates h/c locally. Per step the only global
// traffic is the h vectors (896 bf16) exchanged at device scope plus three
// sense-reversing grid barriers. Gate GEMVs use v_wmma_f32_16x16x32_bf16
// with h replicated across the 16 N columns.
// ---------------------------------------------------------------------------

typedef __attribute__((ext_vector_type(16))) __bf16 v16bf;
typedef __attribute__((ext_vector_type(8)))  float  v8f;

#define SEQ_LEN 32768
#define FR 512            // bf16 elems per 16x32 A-fragment (32 lanes * 16)
#define NBLK 16

// ---- fragment swizzle: (m,k) -> (lane, elem) per ISA 16-bit A 16x32 layout
__global__ void convert_frag_kernel(const float* __restrict__ src,
                                    __bf16* __restrict__ dst, int M, int K) {
  int idx = blockIdx.x * blockDim.x + threadIdx.x;
  if (idx >= M * K) return;
  int m = idx / K, k = idx % K;
  int mtile = m >> 4, kfrag = k >> 5;
  int kp = k & 31;
  int hi = (kp >> 3) & 1;
  int lane = (m & 15) + hi * 16;
  int e = (kp & 16) ? (8 + (kp & 7)) : (kp & 7);
  int nkf = K >> 5;
  size_t off = (((size_t)(mtile * nkf + kfrag)) * 32 + lane) * 16 + e;
  dst[off] = (__bf16)src[idx];
}

__global__ void prep_bias_kernel(const float* __restrict__ x,
                                 const float* __restrict__ Wih1,
                                 const float* __restrict__ bih1, const float* __restrict__ bhh1,
                                 const float* __restrict__ bih2, const float* __restrict__ bhh2,
                                 const float* __restrict__ bih3, const float* __restrict__ bhh3,
                                 float* __restrict__ xp1, float* __restrict__ b2,
                                 float* __restrict__ b3) {
  int i = blockIdx.x * blockDim.x + threadIdx.x;
  if (i < 512) {
    float s = bih1[i] + bhh1[i];
    #pragma unroll 8
    for (int k = 0; k < 128; ++k) s += Wih1[i * 128 + k] * x[k];
    xp1[i] = s;
  }
  if (i < 1024) b2[i] = bih2[i] + bhh2[i];
  if (i < 2048) b3[i] = bih3[i] + bhh3[i];
}

__global__ void init_sync_kernel(int* counter, int* gen) {
  if (threadIdx.x == 0 && blockIdx.x == 0) { *counter = 0; *gen = 0; }
}

// ---------------------------------------------------------------------------
__device__ __forceinline__ float sigf(float x) { return 1.0f / (1.0f + __expf(-x)); }

// acc += W_tile(16 x 32*nkf) @ h, h replicated across all 16 N columns.
__device__ __forceinline__ v8f wmma_gemv(v8f acc, const __bf16* fragbase,
                                         const __bf16* hlds, int nkf, int lane) {
  const int koff = (lane & 16) ? 8 : 0;
  for (int kf = 0; kf < nkf; ++kf) {
    v16bf a = *(const v16bf*)(fragbase + ((size_t)kf * 32 + lane) * 16);
    v16bf b;
    const int base = kf * 32;
    #pragma unroll
    for (int e = 0; e < 16; ++e) {
      int k = base + ((e < 8) ? (koff + e) : (16 + koff + (e - 8)));
      b[e] = hlds[k];
    }
    acc = __builtin_amdgcn_wmma_f32_16x16x32_bf16(false, a, false, b,
                                                  (short)0, acc, false, false);
  }
  return acc;
}

// grid sense barrier across NBLK persistent blocks (device-scope atomics)
__device__ __forceinline__ void gbar(int* counter, int* gen, int tid, int& mygen) {
  __syncthreads();
  if (tid == 0) {
    __threadfence();
    int target = ++mygen;
    if (atomicAdd(counter, 1) == NBLK - 1) {
      __hip_atomic_store(counter, 0, __ATOMIC_RELAXED, __HIP_MEMORY_SCOPE_AGENT);
      __hip_atomic_store(gen, target, __ATOMIC_RELEASE, __HIP_MEMORY_SCOPE_AGENT);
    } else {
      while (__hip_atomic_load(gen, __ATOMIC_ACQUIRE, __HIP_MEMORY_SCOPE_AGENT) < target)
        __builtin_amdgcn_s_sleep(2);
    }
  }
  __syncthreads();
}

__device__ __forceinline__ void store_hslice(unsigned int* hgw, int idx, float hval) {
  __bf16 hv = (__bf16)hval;
  unsigned short bits = __builtin_bit_cast(unsigned short, hv);
  __hip_atomic_store((unsigned short*)hgw + idx, bits,
                     __ATOMIC_RELAXED, __HIP_MEMORY_SCOPE_AGENT);
}

// ---------------------------------------------------------------------------
__global__ __launch_bounds__(256) void lstm3_shard_kernel(
    const __bf16* __restrict__ Whh1f, const __bf16* __restrict__ Wih2f,
    const __bf16* __restrict__ Whh2f, const __bf16* __restrict__ Wih3f,
    const __bf16* __restrict__ Whh3f, const float* __restrict__ xp1g,
    const float* __restrict__ b2g, const float* __restrict__ b3g,
    const float* __restrict__ Wout, const float* __restrict__ boutp,
    int* bar_counter, int* bar_gen,
    unsigned int* hg1, unsigned int* hg2, unsigned int* hg3,
    float* __restrict__ out) {
  const int tid  = threadIdx.x;
  const int b    = blockIdx.x;
  const int wave = tid >> 5;
  const int lane = tid & 31;

  // ---- LDS: weight shard (dynamic, 256 KB) + small state -----------------
  extern __shared__ __bf16 wlds[];
  __bf16* W1  = wlds;             // 16 frags (blocks 0..7 only)
  __bf16* W2i = wlds + 16 * FR;   // 16 frags
  __bf16* W2h = wlds + 32 * FR;   // 32 frags
  __bf16* W3i = wlds + 64 * FR;   // 64 frags
  __bf16* W3h = wlds + 128 * FR;  // 128 frags

  __shared__ unsigned int h1w[64];   // h1 replica, 128 bf16
  __shared__ unsigned int h2w[128];  // h2 replica, 256 bf16
  __shared__ unsigned int h3w[256];  // h3 replica, 512 bf16
  __bf16* h1 = (__bf16*)h1w;
  __bf16* h2 = (__bf16*)h2w;
  __bf16* h3 = (__bf16*)h3w;
  __shared__ float c1s[16], c2s[16], c3s[32];      // private cell slices
  __shared__ float bias1[64], bias2[64], bias3[128];
  __shared__ float g12[64], g3[128];               // gate slices
  __shared__ float redbuf;

  // ---- one-time: preload weight shard global -> LDS ----------------------
  // each frag = 256 u32 words; 256 threads copy one word each
  if (b < 8) {
    for (int f = 0; f < 16; ++f) {                  // layer1 Whh1
      int gm = f >> 2, kf = f & 3, mt = gm * 8 + b;
      ((unsigned int*)(W1 + f * FR))[tid] =
          ((const unsigned int*)(Whh1f + (size_t)(mt * 4 + kf) * FR))[tid];
    }
  }
  for (int f = 0; f < 16; ++f) {                    // layer2 W_ih2
    int gm = f >> 2, kf = f & 3, mt = gm * 16 + b;
    ((unsigned int*)(W2i + f * FR))[tid] =
        ((const unsigned int*)(Wih2f + (size_t)(mt * 4 + kf) * FR))[tid];
  }
  for (int f = 0; f < 32; ++f) {                    // layer2 W_hh2
    int gm = f >> 3, kf = f & 7, mt = gm * 16 + b;
    ((unsigned int*)(W2h + f * FR))[tid] =
        ((const unsigned int*)(Whh2f + (size_t)(mt * 8 + kf) * FR))[tid];
  }
  for (int f = 0; f < 64; ++f) {                    // layer3 W_ih3
    int mtl = f >> 3, kf = f & 7;
    int mt = (mtl >> 1) * 32 + 2 * b + (mtl & 1);
    ((unsigned int*)(W3i + f * FR))[tid] =
        ((const unsigned int*)(Wih3f + (size_t)(mt * 8 + kf) * FR))[tid];
  }
  for (int f = 0; f < 128; ++f) {                   // layer3 W_hh3
    int mtl = f >> 4, kf = f & 15;
    int mt = (mtl >> 1) * 32 + 2 * b + (mtl & 1);
    ((unsigned int*)(W3h + f * FR))[tid] =
        ((const unsigned int*)(Whh3f + (size_t)(mt * 16 + kf) * FR))[tid];
  }

  // ---- one-time: biases (own gate rows), zero state ----------------------
  if (b < 8 && tid < 64) { int g = tid >> 4, r = tid & 15; bias1[tid] = xp1g[g * 128 + b * 16 + r]; }
  if (tid < 64)  { int g = tid >> 4, r = tid & 15; bias2[tid] = b2g[g * 256 + b * 16 + r]; }
  if (tid < 128) { int g = tid >> 5, j = tid & 31; bias3[tid] = b3g[g * 512 + b * 32 + j]; }
  if (tid < 64)  h1w[tid] = 0u;
  if (tid < 128) h2w[tid] = 0u;
  h3w[tid] = 0u;
  if (tid < 16) { c1s[tid] = 0.0f; c2s[tid] = 0.0f; }
  if (tid < 32) c3s[tid] = 0.0f;
  if (tid == 0) redbuf = 0.0f;
  __syncthreads();

  const float bo  = boutp[0];
  const float wv0 = (b == 0) ? Wout[tid]       : 0.0f;
  const float wv1 = (b == 0) ? Wout[tid + 256] : 0.0f;
  int mygen = 0;

  for (int t = 0; t < SEQ_LEN; ++t) {
    asm volatile("" ::: "memory");   // keep LDS fragment loads inside the loop

    // ==== layer 1: blocks 0..7, waves 0..3 own one 16-row gate tile =======
    if (b < 8 && wave < 4) {
      v8f acc = {};
      acc = wmma_gemv(acc, W1 + (size_t)wave * 4 * FR, h1, 4, lane);
      if ((lane & 15) == 0) {
        int mb = wave * 16 + ((lane & 16) ? 8 : 0);
        #pragma unroll
        for (int r = 0; r < 8; ++r) g12[mb + r] = acc[r];
      }
    }
    __syncthreads();
    if (b < 8 && tid < 16) {
      int j = tid;
      float iv = sigf(g12[j] + bias1[j]);
      float fv = sigf(g12[16 + j] + bias1[16 + j]);
      float gv = tanhf(g12[32 + j] + bias1[32 + j]);
      float ov = sigf(g12[48 + j] + bias1[48 + j]);
      float cn = fv * c1s[j] + iv * gv;
      c1s[j] = cn;
      store_hslice(hg1, b * 16 + j, ov * tanhf(cn));
    }
    gbar(bar_counter, bar_gen, tid, mygen);
    if (tid < 64)
      h1w[tid] = __hip_atomic_load(hg1 + tid, __ATOMIC_RELAXED, __HIP_MEMORY_SCOPE_AGENT);
    __syncthreads();

    // ==== layer 2: all blocks, waves 0..3 =================================
    if (wave < 4) {
      v8f acc = {};
      acc = wmma_gemv(acc, W2i + (size_t)wave * 4 * FR, h1, 4, lane);
      acc = wmma_gemv(acc, W2h + (size_t)wave * 8 * FR, h2, 8, lane);
      if ((lane & 15) == 0) {
        int mb = wave * 16 + ((lane & 16) ? 8 : 0);
        #pragma unroll
        for (int r = 0; r < 8; ++r) g12[mb + r] = acc[r];
      }
    }
    __syncthreads();
    if (tid < 16) {
      int j = tid;
      float iv = sigf(g12[j] + bias2[j]);
      float fv = sigf(g12[16 + j] + bias2[16 + j]);
      float gv = tanhf(g12[32 + j] + bias2[32 + j]);
      float ov = sigf(g12[48 + j] + bias2[48 + j]);
      float cn = fv * c2s[j] + iv * gv;
      c2s[j] = cn;
      store_hslice(hg2, b * 16 + j, ov * tanhf(cn));
    }
    gbar(bar_counter, bar_gen, tid, mygen);
    if (tid < 128)
      h2w[tid] = __hip_atomic_load(hg2 + tid, __ATOMIC_RELAXED, __HIP_MEMORY_SCOPE_AGENT);
    __syncthreads();

    // ==== layer 3: all blocks, all 8 waves ================================
    {
      int g = wave >> 1, half = wave & 1;
      v8f acc = {};
      acc = wmma_gemv(acc, W3i + (size_t)wave * 8 * FR, h2, 8, lane);
      acc = wmma_gemv(acc, W3h + (size_t)wave * 16 * FR, h3, 16, lane);
      if ((lane & 15) == 0) {
        int mb = g * 32 + half * 16 + ((lane & 16) ? 8 : 0);
        #pragma unroll
        for (int r = 0; r < 8; ++r) g3[mb + r] = acc[r];
      }
    }
    __syncthreads();
    if (tid < 32) {
      int j = tid;
      float iv = sigf(g3[j] + bias3[j]);
      float fv = sigf(g3[32 + j] + bias3[32 + j]);
      float gv = tanhf(g3[64 + j] + bias3[64 + j]);
      float ov = sigf(g3[96 + j] + bias3[96 + j]);
      float cn = fv * c3s[j] + iv * gv;
      c3s[j] = cn;
      store_hslice(hg3, b * 32 + j, ov * tanhf(cn));
    }
    gbar(bar_counter, bar_gen, tid, mygen);
    h3w[tid] = __hip_atomic_load(hg3 + tid, __ATOMIC_RELAXED, __HIP_MEMORY_SCOPE_AGENT);
    __syncthreads();

    // ==== output: block 0 only, out[t] = dot(W_out, h3) + b_out ===========
    if (b == 0) {
      float p = wv0 * (float)h3[tid] + wv1 * (float)h3[tid + 256];
      #pragma unroll
      for (int off = 16; off > 0; off >>= 1) p += __shfl_down(p, off, 32);
      if (lane == 0) atomicAdd(&redbuf, p);
      __syncthreads();
      if (tid == 0) { out[t] = redbuf + bo; redbuf = 0.0f; }
    }
  }
}

// ---------------------------------------------------------------------------
extern "C" void kernel_launch(void* const* d_in, const int* in_sizes, int n_in,
                              void* d_out, int out_size, void* d_ws, size_t ws_size,
                              hipStream_t stream) {
  (void)in_sizes; (void)n_in; (void)out_size; (void)ws_size;
  const float* x    = (const float*)d_in[0];
  const float* Wih1 = (const float*)d_in[1];
  const float* Whh1 = (const float*)d_in[2];
  const float* bih1 = (const float*)d_in[3];
  const float* bhh1 = (const float*)d_in[4];
  const float* Wih2 = (const float*)d_in[5];
  const float* Whh2 = (const float*)d_in[6];
  const float* bih2 = (const float*)d_in[7];
  const float* bhh2 = (const float*)d_in[8];
  const float* Wih3 = (const float*)d_in[9];
  const float* Whh3 = (const float*)d_in[10];
  const float* bih3 = (const float*)d_in[11];
  const float* bhh3 = (const float*)d_in[12];
  const float* Wout = (const float*)d_in[13];
  const float* bout = (const float*)d_in[14];
  float* out = (float*)d_out;

  // workspace: bf16 fragment weights | f32 bias vectors | sync | h exchange
  __bf16* wsb   = (__bf16*)d_ws;
  __bf16* Whh1f = wsb;                    // 65536
  __bf16* Wih2f = Whh1f + 512 * 128;      // 131072
  __bf16* Whh2f = Wih2f + 1024 * 128;     // 262144
  __bf16* Wih3f = Whh2f + 1024 * 256;     // 524288
  __bf16* Whh3f = Wih3f + 2048 * 256;     // 1048576
  float*  wsf   = (float*)(Whh3f + 2048 * 512);
  float*  xp1   = wsf;                    // 512
  float*  b2f   = wsf + 512;              // 1024
  float*  b3f   = wsf + 1536;             // 2048
  int* ibase = (int*)(wsf + 3584);
  int* bar_counter = ibase;
  int* bar_gen     = ibase + 1;
  unsigned int* hg1 = (unsigned int*)(ibase + 4);  // 64 u32  (128 bf16)
  unsigned int* hg2 = hg1 + 64;                    // 128 u32 (256 bf16)
  unsigned int* hg3 = hg2 + 128;                   // 256 u32 (512 bf16)

  auto conv = [&](const float* src, __bf16* dst, int M, int K) {
    int n = M * K;
    convert_frag_kernel<<<(n + 255) / 256, 256, 0, stream>>>(src, dst, M, K);
  };
  conv(Whh1, Whh1f, 512, 128);
  conv(Wih2, Wih2f, 1024, 128);
  conv(Whh2, Whh2f, 1024, 256);
  conv(Wih3, Wih3f, 2048, 256);
  conv(Whh3, Whh3f, 2048, 512);

  prep_bias_kernel<<<8, 256, 0, stream>>>(x, Wih1, bih1, bhh1, bih2, bhh2,
                                          bih3, bhh3, xp1, b2f, b3f);
  init_sync_kernel<<<1, 32, 0, stream>>>(bar_counter, bar_gen);

  lstm3_shard_kernel<<<NBLK, 256, 256 * FR * (int)sizeof(__bf16), stream>>>(
      Whh1f, Wih2f, Whh2f, Wih3f, Whh3f, xp1, b2f, b3f, Wout, bout,
      bar_counter, bar_gen, hg1, hg2, hg3, out);
}
